// FastLearnableEMA_6433861009532
// MI455X (gfx1250) — matrix operand order repaired and checked by reading, not compiled, not run on hardware
//
#include <hip/hip_runtime.h>
#include <hip/hip_fp16.h>

// ---------------------------------------------------------------------------
// FastLearnableEMA on MI455X (gfx1250).
// x:(32,4096,256) f32, logit_alpha:(256,) f32 -> y:(32,4096,256) f32.
// Memory-bound: 268 MB min traffic ~= 11.5 us @ 23.3 TB/s. Three-phase chunked
// scan; phase 3 re-read of x is L2-resident (134 MB < 192 MB L2).
// Phase 3 uses WMMA (prefix = LowerTri @ Z) + async global->LDS staging.
// ---------------------------------------------------------------------------

#define B_   32
#define T_   4096
#define C_   256
#define K16_ (T_ / 16)      // 256 chunks of 16 timesteps
#define CLAMP_LO_ 1e-4f
#define CLAMP_HI_ (1.0f - 1e-4f)
#define DENOM_EPS_ 1e-8f

typedef __attribute__((ext_vector_type(16))) _Float16 v16h;
typedef __attribute__((ext_vector_type(8)))  float    v8f;
typedef __attribute__((ext_vector_type(4)))  int      v4i;

typedef __attribute__((address_space(1))) v4i* gv4i_ptr;
typedef __attribute__((address_space(3))) v4i* lv4i_ptr;

#if __has_builtin(__builtin_amdgcn_global_load_async_to_lds_b128)
#define EMA_ASYNC 1
#endif

__device__ __forceinline__ float ema_alpha(float la) {
  float a = 1.0f / (1.0f + __expf(-la));
  return fminf(fmaxf(a, CLAMP_LO_), CLAMP_HI_);
}

// ---------------------------------------------------------------------------
// Phase 1: per (b, 64-timestep slab), one thread per channel (coalesced 1KB
// rows). Emit local EMA end-state of each 16-step chunk:
//   part[b,k,c] = sum_{j=0..15} a^(15-j) x[b, 16k+j, c]
// ---------------------------------------------------------------------------
__global__ void __launch_bounds__(256)
ema_phase1(const float* __restrict__ x, const float* __restrict__ logit_alpha,
           float* __restrict__ part) {
  const int blk = blockIdx.x;       // b * (T/64) + slab
  const int b    = blk >> 6;        // T/64 == 64
  const int slab = blk & 63;
  const int c    = threadIdx.x;
  const float a  = ema_alpha(logit_alpha[c]);
  const float* p = x + ((size_t)b * T_ + (size_t)slab * 64) * C_ + c;
  const int kk   = b * K16_ + slab * 4;
  float s = 0.0f;
#pragma unroll
  for (int j = 0; j < 64; ++j) {
    s = fmaf(a, s, p[(size_t)j * C_]);
    if ((j & 15) == 15) { part[(size_t)(kk + (j >> 4)) * C_ + c] = s; s = 0.0f; }
  }
}

// ---------------------------------------------------------------------------
// Phase 2: in-place serial carry scan over the 256 chunks per (b,c):
//   carry[k] = S_end[k-1],  S_end[k] = a^16 * S_end[k-1] + part[k]
// 16 MB of perfectly coalesced traffic; short dependent-FMA chain.
// ---------------------------------------------------------------------------
__global__ void __launch_bounds__(256)
ema_phase2(const float* __restrict__ logit_alpha, float* __restrict__ part) {
  const int b = blockIdx.x;
  const int c = threadIdx.x;
  const float a    = ema_alpha(logit_alpha[c]);
  const float a16  = __expf(16.0f * __logf(a));
  float carry = 0.0f;
  float* p = part + (size_t)b * K16_ * C_ + c;
  for (int k = 0; k < K16_; ++k) {
    float v = p[(size_t)k * C_];
    p[(size_t)k * C_] = carry;          // carry INTO chunk k
    carry = fmaf(a16, carry, v);
  }
}

// ---------------------------------------------------------------------------
// Phase 3: per (b, 16-step tile): stage tile to LDS (async b128), then per
// wave compute two 16-channel groups via one WMMA each:
//   P = Ltri(16x16 ones, f16-exact) @ Z,  Z[j,n] = x[j,c]*a_c^-j
//   S[m] = a^m * (P[m] + a*carry);  y = (1-a)*S / max(1-a^t, eps)
// A/B fragments use contiguous k = 16*h + s order, not the ISA table order (K=16..31 padded
// with zeros -> hi-lane B is zero, hi-lane A is zero).
// ---------------------------------------------------------------------------
__global__ void __launch_bounds__(256)
ema_phase3(const float* __restrict__ x, const float* __restrict__ logit_alpha,
           const float* __restrict__ carryws, float* __restrict__ y) {
  __shared__ float sx[16 * C_];                 // 16 KB tile: [t_local][c]
  const int blk = blockIdx.x;                   // b * K16 + k
  const int b   = blk >> 8;                     // K16 == 256
  const int k   = blk & 255;
  const int t0  = k * 16;
  const int tid = threadIdx.x;
  const size_t xbase = ((size_t)b * T_ + t0) * C_;

#ifdef EMA_ASYNC
  {
    const float* gsrc = x + xbase;
#pragma unroll
    for (int i = 0; i < 4; ++i) {
      const int off = (tid + i * 256) * 4;      // 16B (b128) granules
      __builtin_amdgcn_global_load_async_to_lds_b128(
          (gv4i_ptr)(gsrc + off),
          (lv4i_ptr)(&sx[off]),
          0, 0);
    }
#if __has_builtin(__builtin_amdgcn_s_wait_asynccnt)
    __builtin_amdgcn_s_wait_asynccnt(0);
#else
    asm volatile("s_wait_asynccnt 0" ::: "memory");
#endif
  }
  __syncthreads();
#else
  {
    const float4* gsrc = (const float4*)(x + xbase);
    float4* dst = (float4*)sx;
#pragma unroll
    for (int i = 0; i < 4; ++i) dst[tid + i * 256] = gsrc[tid + i * 256];
  }
  __syncthreads();
#endif

  const int wave = tid >> 5;
  const int lane = tid & 31;
  const int n    = lane & 15;
  const bool hi  = lane >= 16;

  // A fragment: 16x32 f16; lane = M (= lane%16); slots 0-15 hold K = lo?0-15:16-31
  // (hi lanes are zero padding). Ltri[m,k] = (k<=m).
  v16h afrag;
  {
    const int m = n;
#pragma unroll
    for (int s = 0; s < 16; ++s) {
      const int kk = s + (hi ? 16 : 0);
      afrag[s]     = (_Float16)((kk <= m) ? 1.0f : 0.0f);
    }
  }

#pragma unroll
  for (int g = 0; g < 2; ++g) {
    const int cbase = (wave * 2 + g) * 16;
    const int c     = cbase + n;
    const float a    = ema_alpha(logit_alpha[c]);
    const float loga = __logf(a);
    const float ainv = __expf(-loga);

    // B fragment: 32x16 f16; lane = N; lo lanes hold K=0..15 (Z column),
    // hi lanes hold K=16..31 (zeros).
    v16h bfrag;
    {
      float r = 1.0f;
      const float mask = hi ? 0.0f : 1.0f;
#pragma unroll
      for (int j = 0; j < 16; ++j) {
        bfrag[j] = (_Float16)(sx[j * C_ + c] * r * mask);
        r *= ainv;
      }
    }

    v8f acc = {};
    acc = __builtin_amdgcn_wmma_f32_16x16x32_f16(
        /*neg_a=*/false, afrag, /*neg_b=*/false, bfrag,
        /*c_mod=*/(short)0, acc, /*reuse_a=*/false, /*reuse_b=*/false);

    // Epilogue: D layout -> lane holds (m = v + 8*hi, n = lane%16).
    const float carry   = carryws[(size_t)(b * K16_ + k) * C_ + c];
    const float p0      = __expf((float)t0 * loga);   // a^t0
    const float one_m_a = 1.0f - a;
    float am = hi ? __expf(8.0f * loga) : 1.0f;       // a^m start
#pragma unroll
    for (int v = 0; v < 8; ++v) {
      const int m   = v + (hi ? 8 : 0);
      const float s = am * (acc[v] + a * carry);      // a^(m+1)carry + a^m P[m]
      const float d = fmaxf(1.0f - p0 * am, DENOM_EPS_);
      // fast reciprocal (v_rcp_f32) instead of IEEE divide: ~2 ulp, avoids
      // the 8-instruction div_scale/div_fixup chain per output element.
      y[xbase + (size_t)m * C_ + c] = one_m_a * s * __builtin_amdgcn_rcpf(d);
      am *= a;
    }
  }
}

// ---------------------------------------------------------------------------
extern "C" void kernel_launch(void* const* d_in, const int* in_sizes, int n_in,
                              void* d_out, int out_size, void* d_ws, size_t ws_size,
                              hipStream_t stream) {
  (void)in_sizes; (void)n_in; (void)out_size; (void)ws_size;
  const float* x  = (const float*)d_in[0];
  const float* la = (const float*)d_in[1];
  float* y    = (float*)d_out;
  float* part = (float*)d_ws;   // needs B*K16*C floats = 8 MB scratch

  ema_phase1<<<dim3(B_ * (T_ / 64)), dim3(C_), 0, stream>>>(x, la, part);
  ema_phase2<<<dim3(B_), dim3(C_), 0, stream>>>(la, part);
  ema_phase3<<<dim3(B_ * K16_), dim3(256), 0, stream>>>(x, la, part, y);
}
